// MeshGraphDecoder_4552665334031
// MI455X (gfx1250) — compile-verified, matrix operand order checked by reading, not run on hardware
//
#include <hip/hip_runtime.h>
#include <hip/hip_bf16.h>

typedef __attribute__((ext_vector_type(16))) __bf16 v16bf;
typedef __attribute__((ext_vector_type(8)))  float  v8f;

#define LN_EPS 1e-5f

__device__ __forceinline__ v8f wmma_bf16(v16bf a, v16bf b, v8f c) {
  // D = A(16x32 bf16) x B(32x16 bf16) + C(16x16 f32)
  return __builtin_amdgcn_wmma_f32_16x16x32_bf16(false, a, false, b, (short)0, c, false, false);
}

// Load a 16x32 bf16 A-fragment from an LDS row-major tile.
// ISA layout (16-bit A 16x32): lane m = l%16, half g = l/16;
// VGPR v0..3 hold K = g*8 + 2v,2v+1 ; v4..7 hold K = 16 + g*8 + 2(v-4)+{0,1}.
// Each half is 8 consecutive bf16 -> compiler emits 2x ds_load_b128.
__device__ __forceinline__ v16bf load_a_frag(const __bf16* tile, int stride, int lane, int kt) {
  const int m = lane & 15;
  const int g = lane >> 4;
  const __bf16* row = tile + m * stride + kt * 32;
  union { v16bf v; unsigned u[8]; } r;
#pragma unroll
  for (int i = 0; i < 4; ++i) {
    r.u[i]     = *(const unsigned*)(row + g * 8 + 2 * i);
    r.u[i + 4] = *(const unsigned*)(row + 16 + g * 8 + 2 * i);
  }
  return r.v;
}

__device__ __forceinline__ void cvt_store4(__bf16* dst, float4 v) {
  union { __bf16 h[4]; unsigned long long u; } p;
  p.h[0] = (__bf16)v.x; p.h[1] = (__bf16)v.y; p.h[2] = (__bf16)v.z; p.h[3] = (__bf16)v.w;
  *(unsigned long long*)dst = p.u;
}

// Repack W[K,N] (f32, row-major) into bf16 B-fragments: frag (kt,nt) covers
// K rows [kt*32, kt*32+32), N cols [nt*16, nt*16+16). Fragment = 512 bf16,
// element order = lane*16 + v*2 + h so a lane's 16 values are 32 contiguous bytes.
__global__ void pack_w(const float* __restrict__ W, __bf16* __restrict__ out, int K, int N) {
  int idx = blockIdx.x * blockDim.x + threadIdx.x;
  int total = K * N;
  if (idx >= total) return;
  int k = idx / N, n = idx - k * N;
  int kt = k >> 5, kl = k & 31;
  int nt = n >> 4, nl = n & 15;
  int h = kl & 1, g, v;
  if (kl < 8)       { g = 0; v = kl >> 1; }
  else if (kl < 16) { g = 1; v = (kl - 8) >> 1; }
  else if (kl < 24) { g = 0; v = 4 + ((kl - 16) >> 1); }
  else              { g = 1; v = 4 + ((kl - 24) >> 1); }
  int lane = g * 16 + nl;
  size_t frag = (size_t)kt * (N >> 4) + nt;
  out[frag * 512 + lane * 16 + v * 2 + h] = (__bf16)W[idx];
}

__global__ void zero_f(float* __restrict__ p, long n) {
  long i = (long)blockIdx.x * blockDim.x + threadIdx.x;
  long stride = (long)gridDim.x * blockDim.x;
  for (; i < n; i += stride) p[i] = 0.f;
}

// Edge MLP: concat(efeat, mesh[src], grid[dst]) [E,384] -> 256 (SiLU) -> 128 (LN),
// then atomic scatter-sum into agg[dst].
// Block = 128 threads (4 waves), 64 edges. GEMM1: wave owns 4 N-tiles x 4 M-tiles
// (16 independent WMMA chains per K-step). GEMM2: 2 N-tiles x 4 M-tiles.
__global__ __launch_bounds__(128) void edge_mlp_scatter(
    const float* __restrict__ m2g, const float* __restrict__ grid,
    const float* __restrict__ mesh, const int* __restrict__ src,
    const int* __restrict__ dst,
    const __bf16* __restrict__ w1, const float* __restrict__ b1,
    const __bf16* __restrict__ w2, const float* __restrict__ b2,
    const float* __restrict__ gamma, const float* __restrict__ beta,
    float* __restrict__ agg) {
  __shared__ __align__(16) unsigned char smem[64 * 384 * 2];  // 48 KB, phase-aliased
  __bf16* sCat = (__bf16*)smem;   // phase 1: [64][384] bf16 concat inputs
  __bf16* sHid = (__bf16*)smem;   // phase 2: [64][256] bf16 hidden
  float*  sOut = (float*)smem;    // phase 3: [64][128] f32 pre-LN output

  const int tid  = threadIdx.x;
  const int wave = tid >> 5;
  const int lane = tid & 31;
  const int n0   = lane & 15;
  const int g    = lane >> 4;
  const long eBase = (long)blockIdx.x * 64;

  // ---- stage concat tile (float4 gather -> bf16 LDS) ----
  for (int i = tid; i < 64 * 96; i += 128) {   // 96 float4 per row
    int r = i / 96, c4 = i - r * 96;
    long e = eBase + r;
    float4 v;
    if (c4 < 32)      v = ((const float4*)(m2g  + e * 128))[c4];
    else if (c4 < 64) v = ((const float4*)(mesh + (long)src[e] * 128))[c4 - 32];
    else              v = ((const float4*)(grid + (long)dst[e] * 128))[c4 - 64];
    cvt_store4(sCat + r * 384 + c4 * 4, v);
  }
  __syncthreads();

  // ---- GEMM1: [64x384] x [384x256] ----
  v8f acc[4][4];
#pragma unroll
  for (int mt = 0; mt < 4; ++mt)
#pragma unroll
    for (int j = 0; j < 4; ++j) acc[mt][j] = 0.f;
  const v16bf* w1v = (const v16bf*)w1;
#pragma unroll
  for (int kt = 0; kt < 12; ++kt) {
    v16bf a[4], b[4];
#pragma unroll
    for (int mt = 0; mt < 4; ++mt) a[mt] = load_a_frag(sCat + mt * 16 * 384, 384, lane, kt);
#pragma unroll
    for (int j = 0; j < 4; ++j)    b[j]  = w1v[(size_t)(kt * 16 + wave * 4 + j) * 32 + lane];
#pragma unroll
    for (int mt = 0; mt < 4; ++mt)
#pragma unroll
      for (int j = 0; j < 4; ++j)  acc[mt][j] = wmma_bf16(a[mt], b[j], acc[mt][j]);
  }
  __syncthreads();   // all waves done reading sCat

  // ---- bias + SiLU -> hidden tile (bf16, aliases sCat) ----
#pragma unroll
  for (int j = 0; j < 4; ++j) {
    int col = (wave * 4 + j) * 16 + n0;
    float bias = b1[col];
#pragma unroll
    for (int mt = 0; mt < 4; ++mt)
#pragma unroll
      for (int r = 0; r < 8; ++r) {
        float x = acc[mt][j][r] + bias;     // C layout: row = mt*16 + r + 8*g
        x = x / (1.f + __expf(-x));         // SiLU
        sHid[(mt * 16 + r + 8 * g) * 256 + col] = (__bf16)x;
      }
  }
  __syncthreads();

  // ---- GEMM2: [64x256] x [256x128] ----
  v8f acc2[4][2];
#pragma unroll
  for (int mt = 0; mt < 4; ++mt) { acc2[mt][0] = 0.f; acc2[mt][1] = 0.f; }
  const v16bf* w2v = (const v16bf*)w2;
#pragma unroll
  for (int kt = 0; kt < 8; ++kt) {
    v16bf a[4], b[2];
#pragma unroll
    for (int mt = 0; mt < 4; ++mt) a[mt] = load_a_frag(sHid + mt * 16 * 256, 256, lane, kt);
#pragma unroll
    for (int j = 0; j < 2; ++j)    b[j]  = w2v[(size_t)(kt * 8 + wave * 2 + j) * 32 + lane];
#pragma unroll
    for (int mt = 0; mt < 4; ++mt)
#pragma unroll
      for (int j = 0; j < 2; ++j)  acc2[mt][j] = wmma_bf16(a[mt], b[j], acc2[mt][j]);
  }
  __syncthreads();   // all waves done reading sHid

  // ---- bias -> f32 tile in LDS (aliases) for cross-wave LayerNorm ----
#pragma unroll
  for (int j = 0; j < 2; ++j) {
    int col = (wave * 2 + j) * 16 + n0;
    float bias = b2[col];
#pragma unroll
    for (int mt = 0; mt < 4; ++mt)
#pragma unroll
      for (int r = 0; r < 8; ++r)
        sOut[(mt * 16 + r + 8 * g) * 128 + col] = acc2[mt][j][r] + bias;
  }
  __syncthreads();

  // ---- LayerNorm(128) + atomic scatter. 2 lanes per row (64 cols each). ----
  {
    int row = wave * 16 + n0;           // wave owns rows [16w, 16w+16)
    const float* rp = sOut + row * 128 + g * 64;
    float4 xq[16];
    float s = 0.f, s2 = 0.f;
#pragma unroll
    for (int i = 0; i < 16; ++i) {
      xq[i] = ((const float4*)rp)[i];
      s  += xq[i].x + xq[i].y + xq[i].z + xq[i].w;
      s2 += xq[i].x * xq[i].x + xq[i].y * xq[i].y + xq[i].z * xq[i].z + xq[i].w * xq[i].w;
    }
    s  += __shfl_xor(s, 16);            // combine the two half-row lanes
    s2 += __shfl_xor(s2, 16);
    float mean = s * (1.f / 128.f);
    float var  = s2 * (1.f / 128.f) - mean * mean;
    float rstd = rsqrtf(var + LN_EPS);
    long d = dst[eBase + row];
    float* dp = agg + d * 128 + g * 64;
#pragma unroll
    for (int i = 0; i < 16; ++i) {
      int c = g * 64 + i * 4;
      atomicAdd(dp + i * 4 + 0, (xq[i].x - mean) * rstd * gamma[c + 0] + beta[c + 0]);
      atomicAdd(dp + i * 4 + 1, (xq[i].y - mean) * rstd * gamma[c + 1] + beta[c + 1]);
      atomicAdd(dp + i * 4 + 2, (xq[i].z - mean) * rstd * gamma[c + 2] + beta[c + 2]);
      atomicAdd(dp + i * 4 + 3, (xq[i].w - mean) * rstd * gamma[c + 3] + beta[c + 3]);
    }
  }
}

// Node MLP: concat(agg, grid_nfeat) [N,256] -> 256 (SiLU) -> 128 (LN) + residual.
// `out` doubles as agg input and final output (each block reads its rows first).
__global__ __launch_bounds__(128) void node_mlp(
    const float* __restrict__ grid,
    const __bf16* __restrict__ w1, const float* __restrict__ b1,
    const __bf16* __restrict__ w2, const float* __restrict__ b2,
    const float* __restrict__ gamma, const float* __restrict__ beta,
    float* __restrict__ out) {
  __shared__ __align__(16) unsigned char smem[64 * 256 * 2];  // 32 KB, phase-aliased
  __bf16* sCat = (__bf16*)smem;   // [64][256]
  __bf16* sHid = (__bf16*)smem;   // [64][256]
  float*  sOut = (float*)smem;    // [64][128]

  const int tid  = threadIdx.x;
  const int wave = tid >> 5;
  const int lane = tid & 31;
  const int n0   = lane & 15;
  const int g    = lane >> 4;
  const long nBase = (long)blockIdx.x * 64;

  for (int i = tid; i < 64 * 64; i += 128) {   // 64 float4 per row
    int r = i >> 6, c4 = i & 63;
    long node = nBase + r;
    float4 v;
    if (c4 < 32) v = ((const float4*)(out  + node * 128))[c4];        // agg sums
    else         v = ((const float4*)(grid + node * 128))[c4 - 32];
    cvt_store4(sCat + r * 256 + c4 * 4, v);
  }
  __syncthreads();

  // ---- GEMM1: [64x256] x [256x256] ----
  v8f acc[4][4];
#pragma unroll
  for (int mt = 0; mt < 4; ++mt)
#pragma unroll
    for (int j = 0; j < 4; ++j) acc[mt][j] = 0.f;
  const v16bf* w1v = (const v16bf*)w1;
#pragma unroll
  for (int kt = 0; kt < 8; ++kt) {
    v16bf a[4], b[4];
#pragma unroll
    for (int mt = 0; mt < 4; ++mt) a[mt] = load_a_frag(sCat + mt * 16 * 256, 256, lane, kt);
#pragma unroll
    for (int j = 0; j < 4; ++j)    b[j]  = w1v[(size_t)(kt * 16 + wave * 4 + j) * 32 + lane];
#pragma unroll
    for (int mt = 0; mt < 4; ++mt)
#pragma unroll
      for (int j = 0; j < 4; ++j)  acc[mt][j] = wmma_bf16(a[mt], b[j], acc[mt][j]);
  }
  __syncthreads();

#pragma unroll
  for (int j = 0; j < 4; ++j) {
    int col = (wave * 4 + j) * 16 + n0;
    float bias = b1[col];
#pragma unroll
    for (int mt = 0; mt < 4; ++mt)
#pragma unroll
      for (int r = 0; r < 8; ++r) {
        float x = acc[mt][j][r] + bias;
        x = x / (1.f + __expf(-x));
        sHid[(mt * 16 + r + 8 * g) * 256 + col] = (__bf16)x;
      }
  }
  __syncthreads();

  // ---- GEMM2: [64x256] x [256x128] ----
  v8f acc2[4][2];
#pragma unroll
  for (int mt = 0; mt < 4; ++mt) { acc2[mt][0] = 0.f; acc2[mt][1] = 0.f; }
  const v16bf* w2v = (const v16bf*)w2;
#pragma unroll
  for (int kt = 0; kt < 8; ++kt) {
    v16bf a[4], b[2];
#pragma unroll
    for (int mt = 0; mt < 4; ++mt) a[mt] = load_a_frag(sHid + mt * 16 * 256, 256, lane, kt);
#pragma unroll
    for (int j = 0; j < 2; ++j)    b[j]  = w2v[(size_t)(kt * 8 + wave * 2 + j) * 32 + lane];
#pragma unroll
    for (int mt = 0; mt < 4; ++mt)
#pragma unroll
      for (int j = 0; j < 2; ++j)  acc2[mt][j] = wmma_bf16(a[mt], b[j], acc2[mt][j]);
  }
  __syncthreads();

#pragma unroll
  for (int j = 0; j < 2; ++j) {
    int col = (wave * 2 + j) * 16 + n0;
    float bias = b2[col];
#pragma unroll
    for (int mt = 0; mt < 4; ++mt)
#pragma unroll
      for (int r = 0; r < 8; ++r)
        sOut[(mt * 16 + r + 8 * g) * 128 + col] = acc2[mt][j][r] + bias;
  }
  __syncthreads();

  // ---- LayerNorm(128) + residual + store ----
  {
    int row = wave * 16 + n0;
    const float* rp = sOut + row * 128 + g * 64;
    float4 xq[16];
    float s = 0.f, s2 = 0.f;
#pragma unroll
    for (int i = 0; i < 16; ++i) {
      xq[i] = ((const float4*)rp)[i];
      s  += xq[i].x + xq[i].y + xq[i].z + xq[i].w;
      s2 += xq[i].x * xq[i].x + xq[i].y * xq[i].y + xq[i].z * xq[i].z + xq[i].w * xq[i].w;
    }
    s  += __shfl_xor(s, 16);
    s2 += __shfl_xor(s2, 16);
    float mean = s * (1.f / 128.f);
    float var  = s2 * (1.f / 128.f) - mean * mean;
    float rstd = rsqrtf(var + LN_EPS);
    long node = nBase + row;
    const float4* res = (const float4*)(grid + node * 128 + g * 64);
    float4* op = (float4*)(out + node * 128 + g * 64);
#pragma unroll
    for (int i = 0; i < 16; ++i) {
      int c = g * 64 + i * 4;
      float4 rv = res[i];
      float4 y;
      y.x = (xq[i].x - mean) * rstd * gamma[c + 0] + beta[c + 0] + rv.x;
      y.y = (xq[i].y - mean) * rstd * gamma[c + 1] + beta[c + 1] + rv.y;
      y.z = (xq[i].z - mean) * rstd * gamma[c + 2] + beta[c + 2] + rv.z;
      y.w = (xq[i].w - mean) * rstd * gamma[c + 3] + beta[c + 3] + rv.w;
      op[i] = y;
    }
  }
}

extern "C" void kernel_launch(void* const* d_in, const int* in_sizes, int n_in,
                              void* d_out, int out_size, void* d_ws, size_t ws_size,
                              hipStream_t stream) {
  const float* m2g  = (const float*)d_in[0];
  const float* grid = (const float*)d_in[1];
  const float* mesh = (const float*)d_in[2];
  const int*   src  = (const int*)d_in[3];
  const int*   dst  = (const int*)d_in[4];
  const float* eW1  = (const float*)d_in[5];
  const float* eb1  = (const float*)d_in[6];
  const float* eW2  = (const float*)d_in[7];
  const float* eb2  = (const float*)d_in[8];
  const float* eg   = (const float*)d_in[9];
  const float* ebt  = (const float*)d_in[10];
  const float* nW1  = (const float*)d_in[11];
  const float* nb1  = (const float*)d_in[12];
  const float* nW2  = (const float*)d_in[13];
  const float* nb2  = (const float*)d_in[14];
  const float* ng   = (const float*)d_in[15];
  const float* nbt  = (const float*)d_in[16];
  float* out = (float*)d_out;

  const int E  = in_sizes[3];        // 786432
  const int NG = in_sizes[1] / 128;  // 262144

  // Workspace: packed bf16 weights only (~448 KB); d_out doubles as the
  // scatter-sum aggregation buffer so no large scratch is needed.
  __bf16* eW1p = (__bf16*)d_ws;          // 384*256
  __bf16* eW2p = eW1p + 384 * 256;       // 256*128
  __bf16* nW1p = eW2p + 256 * 128;       // 256*256
  __bf16* nW2p = nW1p + 256 * 256;       // 256*128

  pack_w<<<(384 * 256 + 255) / 256, 256, 0, stream>>>(eW1, eW1p, 384, 256);
  pack_w<<<(256 * 128 + 255) / 256, 256, 0, stream>>>(eW2, eW2p, 256, 128);
  pack_w<<<(256 * 256 + 255) / 256, 256, 0, stream>>>(nW1, nW1p, 256, 256);
  pack_w<<<(256 * 128 + 255) / 256, 256, 0, stream>>>(nW2, nW2p, 256, 128);

  zero_f<<<2048, 256, 0, stream>>>(out, (long)NG * 128);

  edge_mlp_scatter<<<E / 64, 128, 0, stream>>>(m2g, grid, mesh, src, dst,
                                               eW1p, eb1, eW2p, eb2, eg, ebt, out);
  node_mlp<<<NG / 64, 128, 0, stream>>>(grid, nW1p, nb1, nW2p, nb2, ng, nbt, out);
}
